// NTXentLoss_38156489458289
// MI455X (gfx1250) — matrix operand order repaired and checked by reading, not compiled, not run on hardware
//
#include <hip/hip_runtime.h>
#include <math.h>

#define NROW   8192     // 2N
#define NCOL   256      // D
#define HALF_N 4096     // N
#define SLICES 8        // column-sweep split per row tile
#define CT_PER_SLICE (NROW / 16 / SLICES)   // 64 column tiles per slice

// exp(sim/temp) = exp(2*sim) = exp2(sim * 2/ln2)
#define TWO_OVER_LN2 2.8853900817779268f

typedef __attribute__((ext_vector_type(16))) _Float16 v16h;
typedef __attribute__((ext_vector_type(8)))  _Float16 v8h;
typedef __attribute__((ext_vector_type(8)))  float    v8f;

// ---------------- Kernel 1: row-normalize z = [zi; zj] -> f16 ----------------
__global__ void ntxent_normalize(const float* __restrict__ zi,
                                 const float* __restrict__ zj,
                                 _Float16* __restrict__ zn) {
    __shared__ float red[256];
    const int row = blockIdx.x;
    const int t   = threadIdx.x;
    const float* src = (row < HALF_N) ? (zi + (size_t)row * NCOL)
                                      : (zj + (size_t)(row - HALF_N) * NCOL);
    float v = src[t];
    red[t] = v * v;
    __syncthreads();
    for (int off = 128; off > 0; off >>= 1) {
        if (t < off) red[t] += red[t + off];
        __syncthreads();
    }
    float norm = fmaxf(sqrtf(red[0]), 1e-8f);
    zn[(size_t)row * NCOL + t] = (_Float16)(v / norm);
}

// ------- Kernel 2: fused sim GEMM + exp + row-sum + positive-pair pick -------
// 4096 waves: 512 row tiles x 8 column slices. Each wave pins its 16x256 f16
// A tile in 64 VGPRs and sweeps 64 column tiles of B with a strided pointer,
// doing 8x v_wmma_f32_16x16x32_f16 + 8 exp2 per tile. All 8 waves of a block
// share one column slice (identical B addresses -> L0 reuse).
__global__ void ntxent_simrow(const _Float16* __restrict__ zn,
                              float* __restrict__ rowsum_part,
                              float* __restrict__ pos) {
    const int lane = threadIdx.x & 31;
    const int lo   = lane & 15;   // row (A) / col (B) within tile
    const int hi   = lane >> 4;   // which K half this lane carries
    // Force wave-uniform values into SGPRs: scalar branches, scalar addressing.
    const int w = __builtin_amdgcn_readfirstlane(
        blockIdx.x * (blockDim.x >> 5) + (threadIdx.x >> 5));
    const int rowTile = w & (NROW / 16 - 1);   // 0..511
    const int slice   = w >> 9;                // 0..7 (uniform within a block)
    const int R   = rowTile * 16;
    const int ct0 = slice * CT_PER_SLICE;
    const int ct1 = ct0 + CT_PER_SLICE;
    const int posTile = ((R + HALF_N) & (NROW - 1)) >> 4;

    // ---- A tile: rows R..R+15, all K. CDNA5 16-bit A layout:
    // lanes 0-15 (hi=0): K = {k0..k0+7, k0+16..k0+23}
    // lanes 16-31(hi=1): K = {k0+8..k0+15, k0+24..k0+31}
    v16h a[8];
    {
        const v8h* arow = (const v8h*)(zn + (size_t)(R + lo) * NCOL);
        #pragma unroll
        for (int kc = 0; kc < 8; ++kc) {
            const int c0 = kc * 4 + hi;          // (kc*32 + 8*hi)/8
            union { v16h v; v8h h[2]; } u;
            u.h[0] = arow[c0];
            u.h[1] = arow[c0 + 2];               // +16 f16
            a[kc] = u.v;
        }
    }

    float rs[8];
    #pragma unroll
    for (int j = 0; j < 8; ++j) rs[j] = 0.0f;

    // Which C element (if any) of the posTile diagonal this lane owns:
    // C layout: VGPR j, lane -> row m = j + 8*hi, col n = lo. Diagonal => lo == m.
    bool diag[8];
    #pragma unroll
    for (int j = 0; j < 8; ++j) diag[j] = (lo == j + 8 * hi);
    float posval = 0.0f;

    // Strided B pointer: one 64-bit add per tile instead of rebuilding the
    // address from ct (stride 16 rows * 512 B = 8 KB, fits the 24-bit offset).
    const v8h* brow = (const v8h*)(zn + (size_t)(ct0 * 16 + lo) * NCOL);

    for (int ct = ct0; ct < ct1; ++ct, brow += 16 * NCOL / 8) {
        // ---- B tile: B = zn^T; column n of B is row (ct*16+n) of zn.
        // CDNA5 16-bit B layout (same as A): lanes 0-15 carry K = {k0..k0+7, k0+16..k0+23},
        // lanes 16-31 carry K = {k0+8..k0+15, k0+24..k0+31}.
        v16h b[8];
        #pragma unroll
        for (int kc = 0; kc < 8; ++kc) {
            const int c0 = kc * 4 + hi;          // (kc*32 + 8*hi)/8
            union { v16h v; v8h h[2]; } u;
            u.h[0] = brow[c0];
            u.h[1] = brow[c0 + 2];
            b[kc] = u.v;
        }

        v8f acc = {};
        #pragma unroll
        for (int kc = 0; kc < 8; ++kc) {
            acc = __builtin_amdgcn_wmma_f32_16x16x32_f16(
                      false, a[kc], false, b[kc], (short)0, acc, false, false);
        }

        float e[8];
        #pragma unroll
        for (int j = 0; j < 8; ++j) {
            e[j] = __builtin_amdgcn_exp2f(acc[j] * TWO_OVER_LN2);
            rs[j] += e[j];
        }

        if (ct == posTile) {                      // scalar branch, rare
            #pragma unroll
            for (int j = 0; j < 8; ++j)
                posval = diag[j] ? e[j] : posval; // v_cndmask, no exec games
        }
    }

    // One predicated store for the positive pair: lane owns row lo iff
    // its half-wave matches the row's half ((lo>>3) == hi), and only the
    // slice that actually swept posTile writes it.
    if (posTile >= ct0 && posTile < ct1 && (lo >> 3) == hi)
        pos[R + lo] = posval;

    // Reduce partial row sums across the 16 lanes (columns) of each half-wave;
    // each (slice,row) pair has a unique writer -> no atomics.
    #pragma unroll
    for (int j = 0; j < 8; ++j) {
        float v = rs[j];
        for (int m = 8; m > 0; m >>= 1) v += __shfl_xor(v, m, 16);
        if (lo == 0) rowsum_part[(size_t)slice * NROW + R + 8 * hi + j] = v;
    }
}

// -------- Kernel 3: mean( log(sum_slices(rowsum) - e^2) - log(pos) ) --------
__global__ void ntxent_finalize(const float* __restrict__ rowsum_part,
                                const float* __restrict__ pos,
                                float* __restrict__ out) {
    __shared__ float red[256];
    const int t = threadIdx.x;
    const float E2 = 7.389056098930650f;   // exp(1/temp) = exp(2)
    float s = 0.0f;
    for (int i = t; i < NROW; i += 256) {
        float rsum = 0.0f;
        #pragma unroll
        for (int sl = 0; sl < SLICES; ++sl)
            rsum += rowsum_part[(size_t)sl * NROW + i];
        s += __logf(rsum - E2) - __logf(pos[i]);
    }
    red[t] = s;
    __syncthreads();
    for (int off = 128; off > 0; off >>= 1) {
        if (t < off) red[t] += red[t + off];
        __syncthreads();
    }
    if (t == 0) out[0] = red[0] / (float)NROW;
}

extern "C" void kernel_launch(void* const* d_in, const int* in_sizes, int n_in,
                              void* d_out, int out_size, void* d_ws, size_t ws_size,
                              hipStream_t stream) {
    (void)in_sizes; (void)n_in; (void)out_size; (void)ws_size;
    const float* zi = (const float*)d_in[0];
    const float* zj = (const float*)d_in[1];

    _Float16* zn   = (_Float16*)d_ws;                                    // 4 MB
    float* rs_part = (float*)((char*)d_ws + (size_t)NROW * NCOL * 2);    // 256 KB
    float* pos     = rs_part + (size_t)SLICES * NROW;                    // 32 KB

    ntxent_normalize<<<NROW, 256, 0, stream>>>(zi, zj, zn);
    // 512 row tiles x 8 slices = 4096 waves = 512 blocks x 8 waves
    ntxent_simrow<<<512, 256, 0, stream>>>(zn, rs_part, pos);
    ntxent_finalize<<<1, 256, 0, stream>>>(rs_part, pos, (float*)d_out);
}
